// DEA_41059887350340
// MI455X (gfx1250) — matrix operand order repaired and checked by reading, not compiled, not run on hardware
//
#include <hip/hip_runtime.h>

// ---------------------------------------------------------------------------
// MI455X (gfx1250) transposed-attention block, bf16 WMMA throughout.
// B=4, C=128, H=W=64, N=4096.
// ---------------------------------------------------------------------------

#define DEVI __device__ __forceinline__

typedef __attribute__((ext_vector_type(16))) __bf16 v16bf;
typedef __attribute__((ext_vector_type(8)))  float  v8f;
typedef __attribute__((ext_vector_type(4)))  int    v4i;

static constexpr int CB  = 4;      // batch
static constexpr int CC  = 128;    // channels
static constexpr int NN  = 4096;   // H*W

struct Frag32B { uint4 lo, hi; };  // 32 bytes = 16 bf16

#if defined(__has_builtin)
#  if __has_builtin(__builtin_amdgcn_global_load_async_to_lds_b128)
#    define HAVE_ASYNC_LDS 1
#  endif
#endif
#ifndef HAVE_ASYNC_LDS
#  define HAVE_ASYNC_LDS 0
#endif

#if HAVE_ASYNC_LDS
typedef __attribute__((address_space(1))) v4i* g_v4i_p;   // global, 64-bit
typedef __attribute__((address_space(3))) v4i* l_v4i_p;   // LDS, 32-bit
#endif

// 16-byte global -> LDS copy. Uses CDNA5 async-to-LDS path when available
// (ASYNCcnt-tracked, no VGPR round trip), else a plain vector copy.
DEVI void async_copy16(unsigned short* lds_dst, const unsigned short* gsrc) {
#if HAVE_ASYNC_LDS
    // Generic LDS address: low 32 bits are the LDS byte offset (ISA aperture rule).
    g_v4i_p gp = (g_v4i_p)(unsigned long long)(size_t)gsrc;
    l_v4i_p lp = (l_v4i_p)(unsigned int)(size_t)lds_dst;
    __builtin_amdgcn_global_load_async_to_lds_b128(gp, lp, 0, 0);
#else
    *(uint4*)lds_dst = *(const uint4*)gsrc;
#endif
}

DEVI void async_join() {
#if HAVE_ASYNC_LDS
#  if __has_builtin(__builtin_amdgcn_s_wait_asynccnt)
    __builtin_amdgcn_s_wait_asynccnt(0);
#  endif
#endif
}

DEVI unsigned short f2bf(float f) {
    unsigned int u = __float_as_uint(f);
    unsigned int r = (u + 0x7FFFu + ((u >> 16) & 1u)) >> 16;  // RNE
    return (unsigned short)r;
}

// A operand: 16x32 (MxK) tile, row-major in LDS with `stride` ushorts.
// lane<16: row=lane,     K in {0..7} u {16..23}
// lane>=16: row=lane-16, K in {8..15} u {24..31}
DEVI v16bf load_frag_a(const unsigned short* base, int stride, int lane) {
    const unsigned short* p = base + (lane & 15) * stride + ((lane >> 4) << 3);
    Frag32B f;
    f.lo = *(const uint4*)(p);
    f.hi = *(const uint4*)(p + 16);
    return __builtin_bit_cast(v16bf, f);
}

// B operand: 32x16 (KxN) tile; LDS holds B^T: [16 n rows][K contiguous].
// lane holds col n=lane&15, K in {8h..8h+7} u {16+8h..23+8h}, h=lane>>4.
DEVI v16bf load_frag_b(const unsigned short* base, int stride, int lane) {
    const unsigned short* p = base + (lane & 15) * stride + ((lane >> 4) << 3);
    Frag32B f;
    f.lo = *(const uint4*)(p);
    f.hi = *(const uint4*)(p + 16);
    return __builtin_bit_cast(v16bf, f);
}

DEVI v8f wmma_bf16(v16bf a, v16bf b, v8f c) {
    return __builtin_amdgcn_wmma_f32_16x16x32_bf16(
        /*neg_a=*/false, a, /*neg_b=*/false, b,
        /*c_mod=*/(short)0, c, /*reuse_a=*/false, /*reuse_b=*/false);
}

// ---------------------------------------------------------------------------
// Kernel 1: xd = (x + rot90(x)) * tril  (same tensor feeds Q and K); also x->bf16
// rot90 (k=1, axes=(2,3)): out[h][w] = x[w][63-h]
// ---------------------------------------------------------------------------
__global__ void k_prep_x(const float* __restrict__ x,
                         unsigned short* __restrict__ xbf,
                         unsigned short* __restrict__ xdbf) {
    int idx = blockIdx.x * blockDim.x + threadIdx.x;     // B*C*H*W threads
    int w  = idx & 63;
    int h  = (idx >> 6) & 63;
    int bc = idx >> 12;
    float xv = x[idx];
    xbf[idx] = f2bf(xv);
    float xr = x[(bc << 12) + (w << 6) + (63 - h)];
    float xd = (h >= w) ? (xv + xr) : 0.0f;
    xdbf[idx] = f2bf(xd);
}

// ---------------------------------------------------------------------------
// Kernel 2: weights -> bf16
// ---------------------------------------------------------------------------
__global__ void k_prep_w(const float* __restrict__ wq, const float* __restrict__ wk,
                         const float* __restrict__ wv, const float* __restrict__ w1,
                         const float* __restrict__ w3,
                         unsigned short* __restrict__ wqb, unsigned short* __restrict__ wkb,
                         unsigned short* __restrict__ wvb, unsigned short* __restrict__ w1b,
                         unsigned short* __restrict__ w3b) {
    int i = blockIdx.x * blockDim.x + threadIdx.x;
    if (i < 16384)                       wqb[i]            = f2bf(wq[i]);
    else if (i < 32768)                  wkb[i - 16384]    = f2bf(wk[i - 16384]);
    else if (i < 49152)                  wvb[i - 32768]    = f2bf(wv[i - 32768]);
    else if (i < 65536)                  w1b[i - 49152]    = f2bf(w1[i - 49152]);
    else if (i < 65536 + 147456)         w3b[i - 65536]    = f2bf(w3[i - 65536]);
}

// ---------------------------------------------------------------------------
// Kernel 3: QKV 1x1 conv as GEMM: O[o,n] = sum_c W[o,c]*S[c,n] + bias[o]
// grid (N/128, 3*B); 256 threads = 8 waves; each WG does [128 o x 128 n].
// ---------------------------------------------------------------------------
__global__ __launch_bounds__(256)
void k_qkv(const unsigned short* __restrict__ xdbf, const unsigned short* __restrict__ xbf,
           const unsigned short* __restrict__ wqb, const unsigned short* __restrict__ wkb,
           const unsigned short* __restrict__ wvb,
           const float* __restrict__ bq, const float* __restrict__ bk,
           const float* __restrict__ bv,
           unsigned short* __restrict__ Q, unsigned short* __restrict__ K,
           unsigned short* __restrict__ V) {
    __shared__ __align__(16) unsigned short Wl[128 * 128];   // [o][c]
    __shared__ __align__(16) unsigned short Xt[128 * 128];   // [n][c] (B^T)
    int n0   = blockIdx.x * 128;
    int head = blockIdx.y >> 2;                              // 0=Q 1=K 2=V
    int b    = blockIdx.y & 3;
    const unsigned short* Wg = head == 0 ? wqb : (head == 1 ? wkb : wvb);
    const unsigned short* S  = (head == 2 ? xbf : xdbf) + (size_t)b * CC * NN;
    const float* bias        = head == 0 ? bq : (head == 1 ? bk : bv);
    unsigned short* O        = (head == 0 ? Q : (head == 1 ? K : V)) + (size_t)b * CC * NN;

    int tid = threadIdx.x;
    for (int i = tid; i < 2048; i += 256)                    // 2048 x 16B chunks
        async_copy16(&Wl[i * 8], &Wg[i * 8]);
    for (int i = tid; i < 128 * 128; i += 256) {
        int nl = i & 127, c = i >> 7;
        Xt[nl * 128 + c] = S[c * NN + n0 + nl];
    }
    async_join();
    __syncthreads();

    int wave = tid >> 5, lane = tid & 31;
    v8f acc[8] = {};
    for (int kk = 0; kk < 128; kk += 32) {
        v16bf a = load_frag_a(&Wl[(wave * 16) * 128 + kk], 128, lane);
        for (int j = 0; j < 8; ++j) {
            v16bf bb = load_frag_b(&Xt[(j * 16) * 128 + kk], 128, lane);
            acc[j] = wmma_bf16(a, bb, acc[j]);
        }
    }
    int col = lane & 15, g = lane >> 4;
    for (int j = 0; j < 8; ++j)
        for (int r = 0; r < 8; ++r) {
            int o = wave * 16 + r + (g << 3);
            O[o * NN + n0 + j * 16 + col] = f2bf(acc[j][r] + bias[o]);
        }
}

// ---------------------------------------------------------------------------
// Kernel 4: softmax row stats (per b,n: max_m S[n,m], sum_m exp)
// S[n,m]=sum_c Q[c,n]K[c,m]. Online merge over m-chunks of 64 — S never hits HBM.
// grid (N/128, B). LDS: Qt 32KB + Kt 16KB + stats.
// ---------------------------------------------------------------------------
__global__ __launch_bounds__(256)
void k_stats(const unsigned short* __restrict__ Q, const unsigned short* __restrict__ K,
             float* __restrict__ rmaxg, float* __restrict__ rinvg) {
    __shared__ __align__(16) unsigned short Qt[128 * 128];   // [n][c]
    __shared__ __align__(16) unsigned short Kt[64 * 128];    // [m][c]
    __shared__ float rmax[128], rsum[128];
    int n0 = blockIdx.x * 128;
    int b  = blockIdx.y;
    const unsigned short* Qb = Q + (size_t)b * CC * NN;
    const unsigned short* Kb = K + (size_t)b * CC * NN;
    int tid = threadIdx.x, wave = tid >> 5, lane = tid & 31;

    for (int i = tid; i < 128 * 128; i += 256) {
        int nl = i & 127, c = i >> 7;
        Qt[nl * 128 + c] = Qb[c * NN + n0 + nl];
    }
    if (tid < 128) { rmax[tid] = -1e30f; rsum[tid] = 0.0f; }
    __syncthreads();

    for (int m0 = 0; m0 < NN; m0 += 64) {
        for (int i = tid; i < 64 * 128; i += 256) {
            int ml = i & 63, c = i >> 6;
            Kt[ml * 128 + c] = Kb[c * NN + m0 + ml];
        }
        __syncthreads();

        v8f acc[4] = {};
        for (int kk = 0; kk < 128; kk += 32) {
            v16bf a = load_frag_a(&Qt[(wave * 16) * 128 + kk], 128, lane);
            for (int j = 0; j < 4; ++j) {
                v16bf bb = load_frag_b(&Kt[(j * 16) * 128 + kk], 128, lane);
                acc[j] = wmma_bf16(a, bb, acc[j]);
            }
        }
        // per-row (n) reductions over m: 16 lanes of a half-group share a row
        for (int r = 0; r < 8; ++r) {
            float mx = fmaxf(fmaxf(acc[0][r], acc[1][r]), fmaxf(acc[2][r], acc[3][r]));
            for (int s = 8; s >= 1; s >>= 1) mx = fmaxf(mx, __shfl_xor(mx, s, 32));
            float se = 0.0f;
            for (int j = 0; j < 4; ++j) se += __expf(acc[j][r] - mx);
            for (int s = 8; s >= 1; s >>= 1) se += __shfl_xor(se, s, 32);
            if ((lane & 15) == r) {   // one writer per row; rows disjoint per wave
                int row = wave * 16 + r + ((lane >> 4) << 3);
                float om = rmax[row], os = rsum[row];
                float nm = fmaxf(om, mx);
                rsum[row] = os * __expf(om - nm) + se * __expf(mx - nm);
                rmax[row] = nm;
            }
        }
        __syncthreads();
    }
    if (tid < 128) {
        rmaxg[b * NN + n0 + tid] = rmax[tid];
        rinvg[b * NN + n0 + tid] = 1.0f / rsum[tid];
    }
}

// ---------------------------------------------------------------------------
// Kernel 5: attn[c,m] = sum_n V[c,n] * exp(S[n,m]-max[n])*inv[n]
// grid (N/128 m-blocks, B). Recompute S in 64-wide n-chunks; P stays in LDS.
// LDS ~81KB (CDNA5 WGP has 320KB).
// ---------------------------------------------------------------------------
__global__ __launch_bounds__(256)
void k_attn(const unsigned short* __restrict__ Q, const unsigned short* __restrict__ K,
            const unsigned short* __restrict__ V,
            const float* __restrict__ rmaxg, const float* __restrict__ rinvg,
            unsigned short* __restrict__ attnbf) {
    __shared__ __align__(16) unsigned short Kt[128 * 128];   // [m][c], fixed per WG
    __shared__ __align__(16) unsigned short Qt[64 * 128];    // [n][c]
    __shared__ __align__(16) unsigned short Vc[128 * 64];    // [c][n]
    __shared__ __align__(16) unsigned short Pt[128 * 64];    // [m][n]  (B^T of P)
    __shared__ float smax[64], sinv[64];
    int m0 = blockIdx.x * 128;
    int b  = blockIdx.y;
    const unsigned short* Qb = Q + (size_t)b * CC * NN;
    const unsigned short* Kb = K + (size_t)b * CC * NN;
    const unsigned short* Vb = V + (size_t)b * CC * NN;
    int tid = threadIdx.x, wave = tid >> 5, lane = tid & 31;

    for (int i = tid; i < 128 * 128; i += 256) {
        int ml = i & 127, c = i >> 7;
        Kt[ml * 128 + c] = Kb[c * NN + m0 + ml];
    }
    v8f acc[8] = {};
    __syncthreads();

    for (int nc = 0; nc < NN; nc += 64) {
        // stage Qt (transposed, scalar), Vc (contiguous rows, async), stats
        for (int i = tid; i < 64 * 128; i += 256) {
            int nl = i & 63, c = i >> 6;
            Qt[nl * 128 + c] = Qb[c * NN + nc + nl];
        }
        for (int i = tid; i < 1024; i += 256) {              // 128c x 8 chunks of 16B
            int c = i >> 3, part = i & 7;
            async_copy16(&Vc[c * 64 + part * 8], &Vb[c * NN + nc + part * 8]);
        }
        if (tid < 64) {
            smax[tid] = rmaxg[b * NN + nc + tid];
            sinv[tid] = rinvg[b * NN + nc + tid];
        }
        async_join();
        __syncthreads();

        // S tiles: 4 n-sub x 8 m-sub = 32 tiles; wave handles 4
        for (int s = 0; s < 4; ++s) {
            int t = wave * 4 + s;
            int nsub = t >> 3, msub = t & 7;
            v8f sa = {};
            for (int kk = 0; kk < 128; kk += 32) {
                v16bf a  = load_frag_a(&Qt[(nsub * 16) * 128 + kk], 128, lane);
                v16bf bb = load_frag_b(&Kt[(msub * 16) * 128 + kk], 128, lane);
                sa = wmma_bf16(a, bb, sa);
            }
            int col = lane & 15, g = lane >> 4;
            for (int r = 0; r < 8; ++r) {
                int nl = nsub * 16 + r + (g << 3);
                float p = __expf(sa[r] - smax[nl]) * sinv[nl];
                Pt[(msub * 16 + col) * 64 + nl] = f2bf(p);
            }
        }
        __syncthreads();

        // attn += V[16c x 64n] * P[64n x 16m]; 2 K-steps per m-subtile
        for (int kk = 0; kk < 64; kk += 32) {
            v16bf a = load_frag_a(&Vc[(wave * 16) * 64 + kk], 64, lane);
            for (int j = 0; j < 8; ++j) {
                v16bf bb = load_frag_b(&Pt[(j * 16) * 64 + kk], 64, lane);
                acc[j] = wmma_bf16(a, bb, acc[j]);
            }
        }
        __syncthreads();
    }

    int col = lane & 15, g = lane >> 4;
    unsigned short* Ab = attnbf + (size_t)b * CC * NN;
    for (int j = 0; j < 8; ++j)
        for (int r = 0; r < 8; ++r) {
            int c = wave * 16 + r + (g << 3);
            Ab[c * NN + m0 + j * 16 + col] = f2bf(acc[j][r]);
        }
}

// ---------------------------------------------------------------------------
// Kernel 6: 3x3 conv as implicit GEMM over k = c*9 + (ky*3+kx)  (matches OIHW),
// fused BN1+ReLU epilogue -> h (bf16). grid (N/128, B).
// ---------------------------------------------------------------------------
__global__ __launch_bounds__(256)
void k_conv3(const unsigned short* __restrict__ attnbf, const unsigned short* __restrict__ w3b,
             const float* __restrict__ bn1g, const float* __restrict__ bn1b,
             unsigned short* __restrict__ hbf) {
    __shared__ __align__(16) unsigned short Al[128 * 32];    // [o][k]
    __shared__ __align__(16) unsigned short Bt[128 * 32];    // [n][k] im2col^T
    int n0 = blockIdx.x * 128;
    int b  = blockIdx.y;
    const unsigned short* Ab = attnbf + (size_t)b * CC * NN;
    int tid = threadIdx.x, wave = tid >> 5, lane = tid & 31;
    v8f acc[8] = {};

    for (int k0 = 0; k0 < 1152; k0 += 32) {
        for (int i = tid; i < 512; i += 256) {               // 128o x 4 chunks of 16B
            int o = i >> 2, part = i & 3;
            async_copy16(&Al[o * 32 + part * 8], &w3b[o * 1152 + k0 + part * 8]);
        }
        for (int i = tid; i < 128 * 32; i += 256) {
            int kl = i & 31, nl = i >> 5;
            int k = k0 + kl;
            int c = k / 9, r = k - c * 9;
            int dy = r / 3 - 1, dx = r % 3 - 1;
            int n = n0 + nl;
            int h = n >> 6, w = n & 63;
            int hh = h + dy, ww = w + dx;
            unsigned short v = 0;
            if ((unsigned)hh < 64u && (unsigned)ww < 64u)
                v = Ab[c * NN + (hh << 6) + ww];
            Bt[nl * 32 + kl] = v;
        }
        async_join();
        __syncthreads();
        v16bf a = load_frag_a(&Al[(wave * 16) * 32], 32, lane);
        for (int j = 0; j < 8; ++j) {
            v16bf bb = load_frag_b(&Bt[(j * 16) * 32], 32, lane);
            acc[j] = wmma_bf16(a, bb, acc[j]);
        }
        __syncthreads();
    }

    const float inv = rsqrtf(1.0f + 1e-5f);
    int col = lane & 15, g = lane >> 4;
    unsigned short* Hb = hbf + (size_t)b * CC * NN;
    for (int j = 0; j < 8; ++j)
        for (int r = 0; r < 8; ++r) {
            int o = wave * 16 + r + (g << 3);
            float y = acc[j][r] * (bn1g[o] * inv) + bn1b[o];
            Hb[o * NN + n0 + j * 16 + col] = f2bf(fmaxf(y, 0.0f));
        }
}

// ---------------------------------------------------------------------------
// Kernel 7: final 1x1 conv + BN2 + ReLU + gamma*h + x (fp32 out). grid (N/128, B).
// ---------------------------------------------------------------------------
__global__ __launch_bounds__(256)
void k_final(const unsigned short* __restrict__ hbf, const unsigned short* __restrict__ w1b,
             const float* __restrict__ bn2g, const float* __restrict__ bn2b,
             const float* __restrict__ gamma, const float* __restrict__ x,
             float* __restrict__ out) {
    __shared__ __align__(16) unsigned short Wl[128 * 128];
    __shared__ __align__(16) unsigned short Xt[128 * 128];
    int n0 = blockIdx.x * 128;
    int b  = blockIdx.y;
    const unsigned short* Hb = hbf + (size_t)b * CC * NN;
    int tid = threadIdx.x, wave = tid >> 5, lane = tid & 31;
    for (int i = tid; i < 2048; i += 256)
        async_copy16(&Wl[i * 8], &w1b[i * 8]);
    for (int i = tid; i < 128 * 128; i += 256) {
        int nl = i & 127, c = i >> 7;
        Xt[nl * 128 + c] = Hb[c * NN + n0 + nl];
    }
    async_join();
    __syncthreads();

    v8f acc[8] = {};
    for (int kk = 0; kk < 128; kk += 32) {
        v16bf a = load_frag_a(&Wl[(wave * 16) * 128 + kk], 128, lane);
        for (int j = 0; j < 8; ++j) {
            v16bf bb = load_frag_b(&Xt[(j * 16) * 128 + kk], 128, lane);
            acc[j] = wmma_bf16(a, bb, acc[j]);
        }
    }
    const float inv = rsqrtf(1.0f + 1e-5f);
    float gm = gamma[0];
    int col = lane & 15, g = lane >> 4;
    for (int j = 0; j < 8; ++j)
        for (int r = 0; r < 8; ++r) {
            int o = wave * 16 + r + (g << 3);
            float y = fmaxf(acc[j][r] * (bn2g[o] * inv) + bn2b[o], 0.0f);
            int idx = (b * CC + o) * NN + n0 + j * 16 + col;
            out[idx] = gm * y + x[idx];
        }
}

// ---------------------------------------------------------------------------
extern "C" void kernel_launch(void* const* d_in, const int* in_sizes, int n_in,
                              void* d_out, int out_size, void* d_ws, size_t ws_size,
                              hipStream_t stream) {
    (void)in_sizes; (void)n_in; (void)out_size; (void)ws_size;
    const float* x    = (const float*)d_in[0];
    const float* wq   = (const float*)d_in[1];
    const float* bq   = (const float*)d_in[2];
    const float* wk   = (const float*)d_in[3];
    const float* bk   = (const float*)d_in[4];
    const float* wv   = (const float*)d_in[5];
    const float* bv   = (const float*)d_in[6];
    const float* gam  = (const float*)d_in[7];
    const float* w3   = (const float*)d_in[8];
    const float* bn1g = (const float*)d_in[9];
    const float* bn1b = (const float*)d_in[10];
    const float* w1   = (const float*)d_in[11];
    const float* bn2g = (const float*)d_in[12];
    const float* bn2b = (const float*)d_in[13];
    float* out = (float*)d_out;

    char* ws = (char*)d_ws;
    size_t off = 0;
    auto alloc = [&](size_t bytes) -> void* {
        void* p = ws + off;
        off = (off + bytes + 255) & ~(size_t)255;
        return p;
    };
    const size_t TEN = (size_t)CB * CC * NN * sizeof(unsigned short);   // 4 MB
    unsigned short* xdbf = (unsigned short*)alloc(TEN);
    unsigned short* xbf  = (unsigned short*)alloc(TEN);
    unsigned short* Q    = (unsigned short*)alloc(TEN);
    unsigned short* K    = (unsigned short*)alloc(TEN);
    unsigned short* V    = (unsigned short*)alloc(TEN);
    unsigned short* attn = (unsigned short*)alloc(TEN);
    unsigned short* hbf  = (unsigned short*)alloc(TEN);
    unsigned short* wqb  = (unsigned short*)alloc(16384 * 2);
    unsigned short* wkb  = (unsigned short*)alloc(16384 * 2);
    unsigned short* wvb  = (unsigned short*)alloc(16384 * 2);
    unsigned short* w1b  = (unsigned short*)alloc(16384 * 2);
    unsigned short* w3b  = (unsigned short*)alloc(147456 * 2);
    float* rmaxg = (float*)alloc((size_t)CB * NN * 4);
    float* rinvg = (float*)alloc((size_t)CB * NN * 4);

    k_prep_x<<<dim3((CB * CC * NN) / 256), dim3(256), 0, stream>>>(x, xbf, xdbf);
    k_prep_w<<<dim3((4 * 16384 + 147456) / 256), dim3(256), 0, stream>>>(
        wq, wk, wv, w1, w3, wqb, wkb, wvb, w1b, w3b);
    k_qkv<<<dim3(NN / 128, 3 * CB), dim3(256), 0, stream>>>(
        xdbf, xbf, wqb, wkb, wvb, bq, bk, bv, Q, K, V);
    k_stats<<<dim3(NN / 128, CB), dim3(256), 0, stream>>>(Q, K, rmaxg, rinvg);
    k_attn<<<dim3(NN / 128, CB), dim3(256), 0, stream>>>(Q, K, V, rmaxg, rinvg, attn);
    k_conv3<<<dim3(NN / 128, CB), dim3(256), 0, stream>>>(attn, w3b, bn1g, bn1b, hbf);
    k_final<<<dim3(NN / 128, CB), dim3(256), 0, stream>>>(hbf, w1b, bn2g, bn2b, gam, x, out);
}
